// PixelMixBlock_V2_2456721293746
// MI455X (gfx1250) — matrix operand order repaired and checked by reading, not compiled, not run on hardware
//
#include <hip/hip_runtime.h>
#include <math.h>

// ---------------------------------------------------------------------------
// Problem constants (fixed by setup_inputs)
// ---------------------------------------------------------------------------
#define kN     32
#define kC     256
#define kHW    784     // 28*28
#define kINTER 128
#define kHin   28
#define kHO    224     // 28*8
#define kWO    224

typedef __bf16 bf16_t;
typedef bf16_t v16bf __attribute__((ext_vector_type(16)));
typedef bf16_t v8bf  __attribute__((ext_vector_type(8)));
typedef float  v8f   __attribute__((ext_vector_type(8)));

// workspace layout (bytes)
#define QBF_BYTES  (kN * kHW * kINTER * 2)        // 6,422,528
#define WQB_BYTES  (kINTER * kC * 2)              // 65,536
#define U_BYTES    (kN * kHW * 4)                 // 100,352

static __device__ __forceinline__ v8f wmma_bf16(v16bf a, v16bf b, v8f c) {
  return __builtin_amdgcn_wmma_f32_16x16x32_bf16(
      /*neg_a=*/false, a, /*neg_b=*/false, b,
      /*c_mod=*/(short)0, c, /*reuse_a=*/false, /*reuse_b=*/false);
}

// ---------------------------------------------------------------------------
// Kernel 0: convert Wq [128][256] fp32 -> bf16 (row-major, contiguous in c)
// ---------------------------------------------------------------------------
__global__ __launch_bounds__(256) void prep_wq(const float* __restrict__ Wq,
                                               bf16_t* __restrict__ Wqb) {
  int id = blockIdx.x * 256 + threadIdx.x;           // 128*256 = 32768 threads
  Wqb[id] = (bf16_t)Wq[id];
}

// ---------------------------------------------------------------------------
// Kernel 1: value embedding u[n][p] = sum_c Wv[c] * X[n][c][p]
// ---------------------------------------------------------------------------
__global__ __launch_bounds__(256) void vembed(const float* __restrict__ X,
                                              const float* __restrict__ Wv,
                                              float* __restrict__ u) {
  int id = blockIdx.x * 256 + threadIdx.x;           // kN*kHW = 25088 threads
  int n = id / kHW, p = id % kHW;
  float s = 0.0f;
  for (int c = 0; c < kC; ++c) s += Wv[c] * X[(n * kC + c) * kHW + p];
  u[id] = s;
}

// ---------------------------------------------------------------------------
// Kernel 2: Q[n][p][i] = sum_c Wq[i][c] * X[n][c][p] + bq[i]  (bf16 out)
// WMMA GEMM: M = p (16-row block), N = i, K = c (256 = 8 x 32)
// grid (49, 32); 128 threads = 4 waves; wave w owns i-tiles w*32 and w*32+16
// ---------------------------------------------------------------------------
__global__ __launch_bounds__(128) void qgemm(const float* __restrict__ X,
                                             const bf16_t* __restrict__ Wqb,
                                             const float* __restrict__ bq,
                                             bf16_t* __restrict__ Qbf) {
  const int n = blockIdx.y;
  const int p0 = blockIdx.x * 16;
  const int tid = threadIdx.x;
  const int w = tid >> 5, lane = tid & 31;
  const int h = lane >> 4, lm = lane & 15;
  const int i0a = w * 32, i0b = w * 32 + 16;
  const int ia = i0a + lm, ib = i0b + lm;
  const int p = p0 + lm;                  // A-row owned by this lane

  v8f acc0 = {}, acc1 = {};
  for (int kb = 0; kb < 8; ++kb) {
    const int k0 = kb * 32;
    // A fragment (16x32, M=p, K=c): lane holds row p; element e -> K = e + 16h (contiguous)
    v16bf a;
#pragma unroll
    for (int e = 0; e < 16; ++e) {
      int c = k0 + e + 16 * h;
      a[e] = (bf16_t)X[(n * kC + c) * kHW + p];
    }
    // B fragment (32x16, K=c, N=i): lane holds col i; element e -> K = e + 16h (contiguous)
    v16bf bA = *(const v16bf*)(Wqb + ia * kC + k0 + 16 * h);
    v16bf bB = *(const v16bf*)(Wqb + ib * kC + k0 + 16 * h);
    acc0 = wmma_bf16(a, bA, acc0);
    acc1 = wmma_bf16(a, bB, acc1);
  }
  // D layout: lane L, vgpr r -> M = r + 8h, N = lm
  const float bqa = bq[ia], bqb = bq[ib];
#pragma unroll
  for (int r = 0; r < 8; ++r) {
    int pr = p0 + r + 8 * h;
    Qbf[(n * kHW + pr) * kINTER + ia] = (bf16_t)(acc0[r] + bqa);
    Qbf[(n * kHW + pr) * kINTER + ib] = (bf16_t)(acc1[r] + bqb);
  }
}

// ---------------------------------------------------------------------------
// Kernel 3: attention. Per workgroup: one (n, 16-row p block).
//   S[16][784] = (Q[n]^T . Q[idx[n]]) / sqrt(128) via WMMA (K=128 = 4x32)
//   then row softmax fused with  mask28[n][p] = sum_c softmax(S)[p][c] * v[c]
// grid (49, 32); 128 threads = 4 waves; wave w owns column tiles w, w+4, ...
// ---------------------------------------------------------------------------
__global__ __launch_bounds__(128) void attn(const bf16_t* __restrict__ Qbf,
                                            const float* __restrict__ u,
                                            const int* __restrict__ index,
                                            const float* __restrict__ Wv,
                                            const float* __restrict__ bv,
                                            const float* __restrict__ lam,
                                            float* __restrict__ mask28) {
  __shared__ float S[16][kHW];     // 50176 B
  __shared__ float Vv[kHW];        //  3136 B
  __shared__ float redm[16][8];
  __shared__ float rowmax[16];
  __shared__ float rede[16][8];
  __shared__ float redv[16][8];

  const int n = blockIdx.y;
  const int p0 = blockIdx.x * 16;
  const int tid = threadIdx.x;
  const int w = tid >> 5, lane = tid & 31;
  const int h = lane >> 4, lm = lane & 15;
  const int nidx = index[n];

  // value vector for this batch: v[c] = u[idx[n]][c] + Wv[C]*(1-lam) + bv
  const float cv = Wv[kC] * (1.0f - lam[0]) + bv[0];
  for (int c = tid; c < kHW; c += 128) Vv[c] = u[nidx * kHW + c] + cv;

  // A fragments for rows p0..p0+15 (reused across all column tiles)
  const bf16_t* Abase = Qbf + (n * kHW + p0 + lm) * kINTER;
  v16bf afr[4];
#pragma unroll
  for (int kb = 0; kb < 4; ++kb) {
    const int k0 = kb * 32;
    union { v16bf v; v8bf hf[2]; } t;
    t.hf[0] = *(const v8bf*)(Abase + k0 + 16 * h);       // K = k0+16h .. +7
    t.hf[1] = *(const v8bf*)(Abase + k0 + 16 * h + 8);   // K = k0+16h+8 .. +7
    afr[kb] = t.v;
  }

  const float scale = 0.0883883476483184f;  // 1/sqrt(128)
  for (int ct = w; ct < 49; ct += 4) {
    const int c0 = ct * 16;
    const bf16_t* Bbase = Qbf + (nidx * kHW + c0 + lm) * kINTER + 16 * h;
    v8f acc = {};
#pragma unroll
    for (int kb = 0; kb < 4; ++kb) {
      v16bf b = *(const v16bf*)(Bbase + kb * 32);        // K = k0+16h .. +15
      acc = wmma_bf16(afr[kb], b, acc);
    }
#pragma unroll
    for (int r = 0; r < 8; ++r) S[r + 8 * h][c0 + lm] = acc[r] * scale;
  }
  __syncthreads();

  // fused softmax . v : 8 threads per row, 98 columns each
  const int row = tid >> 3, seg = tid & 7;
  const int cbeg = seg * 98, cend = cbeg + 98;
  float mx = -1e30f;
  for (int c = cbeg; c < cend; ++c) mx = fmaxf(mx, S[row][c]);
  redm[row][seg] = mx;
  __syncthreads();
  if (seg == 0) {
    float m = redm[row][0];
#pragma unroll
    for (int j = 1; j < 8; ++j) m = fmaxf(m, redm[row][j]);
    rowmax[row] = m;
  }
  __syncthreads();
  const float rm = rowmax[row];
  float se = 0.0f, sv = 0.0f;
  for (int c = cbeg; c < cend; ++c) {
    float e = __expf(S[row][c] - rm);
    se += e;
    sv += e * Vv[c];
  }
  rede[row][seg] = se;
  redv[row][seg] = sv;
  __syncthreads();
  if (seg == 0) {
    float a = 0.0f, b = 0.0f;
#pragma unroll
    for (int j = 0; j < 8; ++j) { a += rede[row][j]; b += redv[row][j]; }
    mask28[n * kHW + p0 + row] = b / a;
  }
}

// ---------------------------------------------------------------------------
// Kernel 4: 8x bilinear upsample (half-pixel centers, edge clamp) + sigmoid,
// write out [n][2][224][224] = cat(1-m, m)
// ---------------------------------------------------------------------------
__global__ __launch_bounds__(256) void upsample_sig(const float* __restrict__ mask28,
                                                    float* __restrict__ out) {
  int id = blockIdx.x * 256 + threadIdx.x;             // kN*kHO*kWO threads
  int x = id % kWO;
  int y = (id / kWO) % kHO;
  int n = id / (kWO * kHO);

  float sx = (x + 0.5f) * 0.125f - 0.5f;
  float sy = (y + 0.5f) * 0.125f - 0.5f;
  sx = fminf(fmaxf(sx, 0.0f), (float)(kHin - 1));
  sy = fminf(fmaxf(sy, 0.0f), (float)(kHin - 1));
  int x0 = (int)sx, y0 = (int)sy;
  int x1 = min(x0 + 1, kHin - 1), y1 = min(y0 + 1, kHin - 1);
  float fx = sx - (float)x0, fy = sy - (float)y0;

  const float* m = mask28 + n * kHW;
  float m00 = m[y0 * kHin + x0], m01 = m[y0 * kHin + x1];
  float m10 = m[y1 * kHin + x0], m11 = m[y1 * kHin + x1];
  float v = (1.0f - fy) * ((1.0f - fx) * m00 + fx * m01) +
            fy * ((1.0f - fx) * m10 + fx * m11);
  float s = 1.0f / (1.0f + __expf(-v));

  out[((size_t)n * 2 + 0) * (kHO * kWO) + y * kWO + x] = 1.0f - s;
  out[((size_t)n * 2 + 1) * (kHO * kWO) + y * kWO + x] = s;
}

// ---------------------------------------------------------------------------
// Launch
// ---------------------------------------------------------------------------
extern "C" void kernel_launch(void* const* d_in, const int* in_sizes, int n_in,
                              void* d_out, int out_size, void* d_ws, size_t ws_size,
                              hipStream_t stream) {
  (void)in_sizes; (void)n_in; (void)out_size; (void)ws_size;
  const float* x   = (const float*)d_in[0];
  const float* lam = (const float*)d_in[1];
  const int*   idx = (const int*)d_in[2];
  // d_in[3] = scale_factor (fixed = 8, baked into upsample kernel)
  const float* Wq  = (const float*)d_in[4];
  const float* bq  = (const float*)d_in[5];
  const float* Wv  = (const float*)d_in[6];
  const float* bv  = (const float*)d_in[7];
  float* out = (float*)d_out;

  char* ws = (char*)d_ws;
  bf16_t* Qbf = (bf16_t*)(ws);
  bf16_t* Wqb = (bf16_t*)(ws + QBF_BYTES);
  float*  u   = (float*)(ws + QBF_BYTES + WQB_BYTES);
  float*  m28 = (float*)(ws + QBF_BYTES + WQB_BYTES + U_BYTES);

  prep_wq<<<(kINTER * kC) / 256, 256, 0, stream>>>(Wq, Wqb);
  vembed<<<(kN * kHW) / 256, 256, 0, stream>>>(x, Wv, u);
  qgemm<<<dim3(kHW / 16, kN), 128, 0, stream>>>(x, Wqb, bq, Qbf);
  attn<<<dim3(kHW / 16, kN), 128, 0, stream>>>(Qbf, u, idx, Wv, bv, lam, m28);
  upsample_sig<<<(kN * kHO * kWO) / 256, 256, 0, stream>>>(m28, out);
}
